// GATLayer_27187142983760
// MI455X (gfx1250) — matrix operand order repaired and checked by reading, not compiled, not run on hardware
//
#include <hip/hip_runtime.h>
#include <hip/hip_bf16.h>

typedef __attribute__((ext_vector_type(16))) _Float16 v16h;
typedef __attribute__((ext_vector_type(8)))  _Float16 v8h;
typedef __attribute__((ext_vector_type(8)))  float    v8f;
typedef __attribute__((ext_vector_type(4)))  int      v4i;

#define N_NODES 8192
#define IN_DIM  256
#define OUT_DIM 128
#define HEADS   2
#define TDIM    256   // OUT_DIM * HEADS

__device__ __forceinline__ float leaky02(float x) { return x > 0.f ? x : 0.2f * x; }

// Order-preserving float<->uint encoding so we can use integer atomicMax on floats.
__device__ __forceinline__ unsigned encf(float v) {
    unsigned b = __float_as_uint(v);
    return (b & 0x80000000u) ? ~b : (b | 0x80000000u);
}
__device__ __forceinline__ float decf(unsigned u) {
    return (u & 0x80000000u) ? __uint_as_float(u & 0x7fffffffu) : __uint_as_float(~u);
}

// ---------------------------------------------------------------------------
// CDNA5 async global->LDS copy (ASYNCcnt-tracked). Signature (probe-confirmed
// by round-2 diagnostic): (v4i AS1* src, v4i AS3* dst, imm offset, imm cpol).
// ---------------------------------------------------------------------------
#if defined(__has_builtin)
#if __has_builtin(__builtin_amdgcn_global_load_async_to_lds_b128)
#define GAT_ASYNC_LDS 1
#endif
#endif

#ifdef GAT_ASYNC_LDS
typedef __attribute__((address_space(1))) v4i as1_v4i;
typedef __attribute__((address_space(3))) v4i as3_v4i;
#endif

__device__ __forceinline__ void stage_row64B_async(const _Float16* g, _Float16* l) {
#ifdef GAT_ASYNC_LDS
    as1_v4i* gp = (as1_v4i*)(const void*)g;
    as3_v4i* lp = (as3_v4i*)(void*)l;
    __builtin_amdgcn_global_load_async_to_lds_b128(gp, lp, 0, 0);
    __builtin_amdgcn_global_load_async_to_lds_b128(gp, lp, 16, 0);
    __builtin_amdgcn_global_load_async_to_lds_b128(gp, lp, 32, 0);
    __builtin_amdgcn_global_load_async_to_lds_b128(gp, lp, 48, 0);
#else
    float4 b0 = *(const float4*)(g);
    float4 b1 = *(const float4*)(g + 8);
    float4 b2 = *(const float4*)(g + 16);
    float4 b3 = *(const float4*)(g + 24);
    *(float4*)(l)      = b0;
    *(float4*)(l + 8)  = b1;
    *(float4*)(l + 16) = b2;
    *(float4*)(l + 24) = b3;
#endif
}

__device__ __forceinline__ void wait_async_le(int n /* 0 or 4, compile-time */) {
#ifdef GAT_ASYNC_LDS
#if __has_builtin(__builtin_amdgcn_s_wait_asynccnt)
    if (n == 0) __builtin_amdgcn_s_wait_asynccnt(0);
    else        __builtin_amdgcn_s_wait_asynccnt(4);
#else
    if (n == 0) asm volatile("s_wait_asynccnt 0x0" ::: "memory");
    else        asm volatile("s_wait_asynccnt 0x4" ::: "memory");
#endif
#else
    (void)n;
#endif
}

// ---------------------------------------------------------------------------
// Kernel 0: reset atomic-max slots (workspace is poisoned with 0xAA).
// ---------------------------------------------------------------------------
__global__ void gat_init(unsigned* __restrict__ gmax) {
    if (threadIdx.x < HEADS) gmax[threadIdx.x] = 0u;  // encoded "most negative"
}

// ---------------------------------------------------------------------------
// Kernel 1: transformed = features @ W^T via v_wmma_f32_16x16x32_f16.
// One wave per 16x16 output tile. Writes fp32 t32[N][256] and f16 tT16[256][N].
// ---------------------------------------------------------------------------
__global__ __launch_bounds__(32) void gat_transform_wmma(
    const float* __restrict__ features, const float* __restrict__ W,
    float* __restrict__ t32, _Float16* __restrict__ tT16)
{
    const int bid  = blockIdx.x;
    const int i0   = (bid >> 4) << 4;   // row tile
    const int n0   = (bid & 15) << 4;   // col tile
    const int lane = threadIdx.x;
    const int lm   = lane & 15;
    const int half = lane >> 4;

    const float* arow = features + (size_t)(i0 + lm) * IN_DIM;  // A: m = lane&15
    const float* brow = W        + (size_t)(n0 + lm) * IN_DIM;  // B: n = lane&15

    v8f acc = {};
    for (int k0 = 0; k0 < IN_DIM; k0 += 32) {
        // A fragment (16-bit A layout: lanes<16 hold K=e(0..7),16..23; lanes>=16 +8)
        const int ka = k0 + (half << 3);
        float4 a0 = *(const float4*)(arow + ka);
        float4 a1 = *(const float4*)(arow + ka + 4);
        float4 a2 = *(const float4*)(arow + ka + 16);
        float4 a3 = *(const float4*)(arow + ka + 20);
        float af[16] = {a0.x,a0.y,a0.z,a0.w, a1.x,a1.y,a1.z,a1.w,
                        a2.x,a2.y,a2.z,a2.w, a3.x,a3.y,a3.z,a3.w};
        v16h A;
        #pragma unroll
        for (int e = 0; e < 16; ++e) A[e] = (_Float16)af[e];

        // B fragment (same K order as A: lanes<16 hold K=e(0..7),16..23; lanes>=16 +8)
        const int kb = k0 + (half << 3);
        float4 b0 = *(const float4*)(brow + kb);
        float4 b1 = *(const float4*)(brow + kb + 4);
        float4 b2 = *(const float4*)(brow + kb + 16);
        float4 b3 = *(const float4*)(brow + kb + 20);
        float bf[16] = {b0.x,b0.y,b0.z,b0.w, b1.x,b1.y,b1.z,b1.w,
                        b2.x,b2.y,b2.z,b2.w, b3.x,b3.y,b3.z,b3.w};
        v16h B;
        #pragma unroll
        for (int e = 0; e < 16; ++e) B[e] = (_Float16)bf[e];

        acc = __builtin_amdgcn_wmma_f32_16x16x32_f16(false, A, false, B,
                                                     (short)0, acc, false, false);
    }

    // C layout: VGPR r -> m = r + 8*(lane>>4), n = lane&15
    #pragma unroll
    for (int r = 0; r < 8; ++r) {
        const int m = i0 + r + (half << 3);
        const int n = n0 + lm;
        t32[(size_t)m * TDIM + n]     = acc[r];
        tT16[(size_t)n * N_NODES + m] = (_Float16)acc[r];
    }
}

// ---------------------------------------------------------------------------
// Kernel 2: per-node src/dst attention scores + global max of dst scores.
// One wave per node (8 waves / block).
// ---------------------------------------------------------------------------
__global__ __launch_bounds__(256) void gat_scores(
    const float* __restrict__ t32, const float* __restrict__ asrc,
    const float* __restrict__ adst, float* __restrict__ sS,
    float* __restrict__ dS, unsigned* __restrict__ gmax)
{
    const int wave = threadIdx.x >> 5, lane = threadIdx.x & 31;
    const int row  = blockIdx.x * 8 + wave;
    const float* tr = t32 + (size_t)row * TDIM;

    float s0 = 0.f, s1 = 0.f, d0 = 0.f, d1 = 0.f;
    #pragma unroll
    for (int it = 0; it < 4; ++it) {
        const int f = lane + 32 * it;
        const float v0 = tr[f], v1 = tr[OUT_DIM + f];
        s0 += v0 * asrc[f];           d0 += v0 * adst[f];
        s1 += v1 * asrc[OUT_DIM + f]; d1 += v1 * adst[OUT_DIM + f];
    }
    #pragma unroll
    for (int m = 16; m > 0; m >>= 1) {
        s0 += __shfl_xor(s0, m, 32); s1 += __shfl_xor(s1, m, 32);
        d0 += __shfl_xor(d0, m, 32); d1 += __shfl_xor(d1, m, 32);
    }
    if (lane == 0) {
        sS[row * 2 + 0] = s0; sS[row * 2 + 1] = s1;
        dS[row * 2 + 0] = d0; dS[row * 2 + 1] = d1;
        atomicMax(&gmax[0], encf(d0));
        atomicMax(&gmax[1], encf(d1));
    }
}

// ---------------------------------------------------------------------------
// Kernel 3: softmax denominators. One 256-thread block per source row.
// Shift M = leaky(s_i + max_j d_j) >= every masked logit (leaky is monotone).
// ---------------------------------------------------------------------------
__global__ __launch_bounds__(256) void gat_denom(
    const int* __restrict__ adj, const float* __restrict__ sS,
    const float* __restrict__ dS, const unsigned* __restrict__ gmax,
    float* __restrict__ Mv, float* __restrict__ rcp)
{
    const int i = blockIdx.x, t = threadIdx.x;
    const float s0 = sS[i * 2 + 0], s1 = sS[i * 2 + 1];
    const float M0 = leaky02(s0 + decf(gmax[0]));
    const float M1 = leaky02(s1 + decf(gmax[1]));

    float a0 = 0.f, a1 = 0.f;
    const int* arow = adj + (size_t)i * N_NODES;
    for (int j = t; j < N_NODES; j += 256) {
        __builtin_prefetch(arow + j + 256, 0, 0);   // global_prefetch_b8
        if (arow[j] > 0) {
            a0 += __expf(leaky02(s0 + dS[j * 2 + 0]) - M0);
            a1 += __expf(leaky02(s1 + dS[j * 2 + 1]) - M1);
        }
    }
    __shared__ float r0[256], r1[256];
    r0[t] = a0; r1[t] = a1; __syncthreads();
    #pragma unroll
    for (int st = 128; st > 0; st >>= 1) {
        if (t < st) { r0[t] += r0[t + st]; r1[t] += r1[t + st]; }
        __syncthreads();
    }
    if (t == 0) {
        Mv[i * 2 + 0] = M0; Mv[i * 2 + 1] = M1;
        rcp[i * 2 + 0] = r0[0] > 0.f ? 1.f / r0[0] : 0.f;  // empty row -> 0 (nan_to_num)
        rcp[i * 2 + 1] = r1[0] > 0.f ? 1.f / r1[0] : 0.f;
    }
}

// ---------------------------------------------------------------------------
// Kernel 4: out = mean_h( softmax(masked logits) @ transformed_h ) via WMMA.
// 128 blocks x 64 rows; 8 waves = (4 m-tiles) x (2 heads); K-chunks of 32.
// bT (V^T tiles) double-buffered via async global->LDS copies:
//   iter c: issue copy for chunk c+1, build exp tile for c (VALU),
//           s_wait_asynccnt<=4 (in-order => chunk c done), barrier, WMMA.
// ---------------------------------------------------------------------------
__global__ __launch_bounds__(256) void gat_aggregate_wmma(
    const int* __restrict__ adj, const _Float16* __restrict__ tT16,
    const float* __restrict__ sS, const float* __restrict__ dS,
    const float* __restrict__ Mv, const float* __restrict__ rcp,
    float* __restrict__ out)
{
    __shared__ __align__(64) _Float16 bT[2][HEADS][OUT_DIM][32]; // double-buffered B
    __shared__ __align__(64) _Float16 pA[HEADS][64][32];         // prob tiles (A)
    __shared__ float sL[64][2], ML[64][2], rL[64][2];
    __shared__ float dchunk[32][2];
    __shared__ float outS[64][OUT_DIM];

    const int tid  = threadIdx.x;
    const int i0   = blockIdx.x * 64;
    const int wave = tid >> 5, lane = tid & 31;
    const int mTile = wave & 3, head = wave >> 2;
    const int half  = lane >> 4, lm = lane & 15;

    if (tid < 128) {
        const int m = tid >> 1, h = tid & 1;
        sL[m][h] = sS[(size_t)(i0 + m) * 2 + h];
        ML[m][h] = Mv[(size_t)(i0 + m) * 2 + h];
        rL[m][h] = rcp[(size_t)(i0 + m) * 2 + h];
    }

    v8f acc[8] = {};

    const int mload = tid >> 2;               // pA staging: row 0..63
    const int kbase = (tid & 3) << 3;         // 8 k's each
    const int bh = tid >> 7, bf = tid & 127;  // bT staging: one f-row each

    const _Float16* bsrc = tT16 + (size_t)tid * N_NODES;  // this thread's V^T row
    const int NCH = N_NODES / 32;

    // Prologue: stage chunk 0 into buffer 0 (async).
    stage_row64B_async(bsrc, &bT[0][bh][bf][0]);

    for (int c = 0; c < NCH; ++c) {
        const int j0  = c * 32;
        const int buf = c & 1;
        __syncthreads();  // compute(c-1) done: pA, dchunk, bT[buf^1] reusable

        if (tid < 64) {
            const int jj = tid >> 1, h = tid & 1;
            dchunk[jj][h] = dS[(size_t)(j0 + jj) * 2 + h];
        }
        if (c + 1 < NCH)  // issue async copy of next chunk into alternate buffer
            stage_row64B_async(bsrc + j0 + 32, &bT[buf ^ 1][bh][bf][0]);
        __syncthreads();  // dchunk visible

        // Stage A: masked exp probabilities (overlaps with in-flight async copy).
        {
            const int* arow = adj + (size_t)(i0 + mload) * N_NODES + j0 + kbase;
            __builtin_prefetch(arow + 32, 0, 0);   // next adjacency chunk
            int4 q0 = *(const int4*)(arow);
            int4 q1 = *(const int4*)(arow + 4);
            const int av[8] = {q0.x,q0.y,q0.z,q0.w, q1.x,q1.y,q1.z,q1.w};
            const float s0v = sL[mload][0], s1v = sL[mload][1];
            const float M0v = ML[mload][0], M1v = ML[mload][1];
            #pragma unroll
            for (int kk = 0; kk < 8; ++kk) {
                const int k = kbase + kk;
                float p0 = 0.f, p1 = 0.f;
                if (av[kk] > 0) {
                    p0 = __expf(leaky02(s0v + dchunk[k][0]) - M0v);
                    p1 = __expf(leaky02(s1v + dchunk[k][1]) - M1v);
                }
                pA[0][mload][k] = (_Float16)p0;
                pA[1][mload][k] = (_Float16)p1;
            }
        }
        // Chunk c's 4 async b128 copies are the oldest outstanding; in-order
        // completion => waiting to <=4 (or 0 on the last chunk) retires them.
        if (c + 1 < NCH) wait_async_le(4);
        else             wait_async_le(0);
        __syncthreads();  // pA + bT[buf] ready for every wave

        // Compute: A frag per ISA layout, then 8 n-tiles of WMMA.
        {
            const _Float16* ap = &pA[head][mTile * 16 + lm][half << 3];
            v8h alo = *(const v8h*)(ap);
            v8h ahi = *(const v8h*)(ap + 16);
            v16h A = __builtin_shufflevector(alo, ahi,
                        0,1,2,3,4,5,6,7,8,9,10,11,12,13,14,15);
            #pragma unroll
            for (int nt = 0; nt < 8; ++nt) {
                const _Float16* bp = &bT[buf][head][nt * 16 + lm][half << 3];
                v8h blo = *(const v8h*)(bp);
                v8h bhi = *(const v8h*)(bp + 16);
                v16h B = __builtin_shufflevector(blo, bhi,
                            0,1,2,3,4,5,6,7,8,9,10,11,12,13,14,15);
                acc[nt] = __builtin_amdgcn_wmma_f32_16x16x32_f16(
                    false, A, false, B, (short)0, acc[nt], false, false);
            }
        }
    }

    __syncthreads();
    // Combine heads: head0 writes, head1 adds; scale by 1/denom and 1/HEADS.
    if (head == 0) {
        #pragma unroll
        for (int nt = 0; nt < 8; ++nt)
            #pragma unroll
            for (int r = 0; r < 8; ++r) {
                const int m = mTile * 16 + r + (half << 3);
                const int f = nt * 16 + lm;
                outS[m][f] = 0.5f * rL[m][0] * acc[nt][r];
            }
    }
    __syncthreads();
    if (head == 1) {
        #pragma unroll
        for (int nt = 0; nt < 8; ++nt)
            #pragma unroll
            for (int r = 0; r < 8; ++r) {
                const int m = mTile * 16 + r + (half << 3);
                const int f = nt * 16 + lm;
                outS[m][f] += 0.5f * rL[m][1] * acc[nt][r];
            }
    }
    __syncthreads();
    for (int idx = tid; idx < 64 * OUT_DIM; idx += 256) {
        const int m = idx >> 7, f = idx & 127;
        out[(size_t)(i0 + m) * OUT_DIM + f] = outS[m][f];
    }
}

// ---------------------------------------------------------------------------
extern "C" void kernel_launch(void* const* d_in, const int* in_sizes, int n_in,
                              void* d_out, int out_size, void* d_ws, size_t ws_size,
                              hipStream_t stream) {
    const float* features  = (const float*)d_in[0];
    const int*   adjacency = (const int*)d_in[1];
    const float* W         = (const float*)d_in[2];
    const float* attn_src  = (const float*)d_in[3];
    const float* attn_dst  = (const float*)d_in[4];
    float* out = (float*)d_out;

    char* ws = (char*)d_ws;
    float*    t32  = (float*)(ws);                                   // 8 MB
    _Float16* tT16 = (_Float16*)(ws + (size_t)N_NODES * TDIM * 4);   // 4 MB
    float*    sS   = (float*)(ws + (size_t)N_NODES * TDIM * 6);
    float*    dS   = sS  + N_NODES * 2;
    float*    Mv   = dS  + N_NODES * 2;
    float*    rcp  = Mv  + N_NODES * 2;
    unsigned* gmax = (unsigned*)(rcp + N_NODES * 2);

    gat_init<<<1, 32, 0, stream>>>(gmax);
    gat_transform_wmma<<<(N_NODES / 16) * (TDIM / 16), 32, 0, stream>>>(
        features, W, t32, tT16);
    gat_scores<<<N_NODES / 8, 256, 0, stream>>>(t32, attn_src, attn_dst,
                                                sS, dS, gmax);
    gat_denom<<<N_NODES, 256, 0, stream>>>(adjacency, sS, dS, gmax, Mv, rcp);
    gat_aggregate_wmma<<<N_NODES / 64, 256, 0, stream>>>(
        adjacency, tT16, sS, dS, Mv, rcp, out);
}
